// MultiheadAttention_31602369364602
// MI455X (gfx1250) — matrix operand order repaired and checked by reading, not compiled, not run on hardware
//
#include <hip/hip_runtime.h>
#include <hip/hip_bf16.h>
#include <stdint.h>

// ---------------- constants from the reference ----------------
#define S_LEN 2048
#define BATCH 2
#define EMB   768
#define NHEAD 12
#define HDIM  64
#define NH_TOT (BATCH*NHEAD)   // 24
#define M_TOT  (S_LEN*BATCH)   // 4096 tokens
#define QK_SCALE 0.125f        // 64^-0.5

typedef unsigned short u16;
typedef unsigned int   u32;
typedef __attribute__((ext_vector_type(16))) __bf16 v16bf;
typedef __attribute__((ext_vector_type(8)))  float  v8f;
typedef __attribute__((ext_vector_type(4)))  u32    u32x4;

struct Frag32 { u32x4 lo, hi; };   // 32 bytes == one v16bf fragment

// fp32 -> bf16 round-to-nearest-even
__device__ inline u16 f2bf(float x) {
  u32 u = __builtin_bit_cast(u32, x);
  u += 0x7fffu + ((u >> 16) & 1u);
  return (u16)(u >> 16);
}

// A-fragment (16x32 bf16, MxK): lane<16 holds K=[k0..k0+7] & [k0+16..k0+23],
// lane>=16 holds K=[k0+8..k0+15] & [k0+24..k0+31]; row = caller-resolved M.
__device__ inline v16bf ld_a(const u16* p, int ld, int row, int k0, int lane) {
  const u16* r = p + (size_t)row * ld + k0 + ((lane >> 4) << 3);
  Frag32 f;
  f.lo = *(const u32x4*)(r);
  f.hi = *(const u32x4*)(r + 16);
  return __builtin_bit_cast(v16bf, f);
}

// B-fragment (32x16 bf16, KxN): lane<16 holds K=[k0..k0+7] & [k0+16..k0+23] for
// column N=lane; lane>=16 holds K=[k0+8..k0+15] & [k0+24..k0+31] for column N=lane-16.
// p is row-major [N][K] (i.e. B^T in memory).
__device__ inline v16bf ld_b(const u16* p, int ld, int col, int k0, int lane) {
  const u16* r = p + (size_t)col * ld + k0 + ((lane >> 4) << 3);
  Frag32 f;
  f.lo = *(const u32x4*)(r);
  f.hi = *(const u32x4*)(r + 16);
  return __builtin_bit_cast(v16bf, f);
}

__device__ inline v8f wmma_bf16(v16bf a, v16bf b, v8f c) {
  return __builtin_amdgcn_wmma_f32_16x16x32_bf16(
      /*neg_a=*/false, a, /*neg_b=*/false, b,
      /*c_mod=*/(short)0, c, /*reuse_a=*/false, /*reuse_b=*/false);
}

// ---------------- kernel 1: fp32 -> bf16 convert ----------------
__global__ void k_cvt(const float* __restrict__ in, u16* __restrict__ out, int n) {
  for (int i = blockIdx.x * blockDim.x + threadIdx.x; i < n;
       i += gridDim.x * blockDim.x)
    out[i] = f2bf(in[i]);
}

// ---------------- kernel 2: fused QKV projection -----------------
// X: [M_TOT][EMB] bf16, W*: [EMB][EMB] bf16 (torch layout [n][e] == B^T)
// z==0 -> Q [nh][s][d] (pre-scaled), z==1 -> K [nh][s][d], z==2 -> V^T [nh][d][s]
__global__ __launch_bounds__(256) void k_qkv(
    const u16* __restrict__ X,
    const u16* __restrict__ Wq, const u16* __restrict__ Wk, const u16* __restrict__ Wv,
    const float* __restrict__ bq, const float* __restrict__ bk, const float* __restrict__ bv,
    u16* __restrict__ Q, u16* __restrict__ K, u16* __restrict__ V) {
  const int lane = threadIdx.x & 31, wave = threadIdx.x >> 5;
  const int n0 = blockIdx.x * 16;
  const int m0 = (blockIdx.y * 8 + wave) * 16;
  const int z  = blockIdx.z;
  const u16*   W    = (z == 0) ? Wq : ((z == 1) ? Wk : Wv);
  const float* bias = (z == 0) ? bq : ((z == 1) ? bk : bv);

  const int l15 = lane & 15;
  const int rA = m0 + l15;
  const int cB = n0 + l15;
  v8f c = {0.f,0.f,0.f,0.f,0.f,0.f,0.f,0.f};
  for (int k0 = 0; k0 < EMB; k0 += 32) {
    v16bf a = ld_a(X, EMB, rA, k0, lane);
    v16bf b = ld_b(W, EMB, cB, k0, lane);
    c = wmma_bf16(a, b, c);
  }
  const float bval = bias[cB];
  const float sc   = (z == 0) ? QK_SCALE : 1.0f;
  const int h = cB >> 6, d = cB & 63;
  const int rbase = m0 + ((lane >> 4) << 3);
#pragma unroll
  for (int r = 0; r < 8; ++r) {
    const int m  = rbase + r;            // token index = s*BATCH + b
    const int s  = m >> 1;
    const int bb = m & 1;
    const int nh = bb * NHEAD + h;
    const u16 val = f2bf((c[r] + bval) * sc);
    if (z == 0)      Q[((size_t)nh * S_LEN + s) * HDIM + d] = val;
    else if (z == 1) K[((size_t)nh * S_LEN + s) * HDIM + d] = val;
    else             V[((size_t)nh * HDIM + d) * S_LEN + s] = val;
  }
}

// ---------------- kernel 3: flash attention ----------------------
// One 16-query tile per wave; online softmax over 32-key chunks.
__global__ __launch_bounds__(256) void k_attn(
    const u16* __restrict__ Qb, const u16* __restrict__ Kb,
    const u16* __restrict__ Vt, const float* __restrict__ bias,
    const unsigned char* __restrict__ mask, u16* __restrict__ attn) {
  __shared__ u16 Pl[8][16 * 32];   // per-wave P staging (D-layout -> A-layout)
  const int lane = threadIdx.x & 31, wave = threadIdx.x >> 5;
  const int nh = blockIdx.x;
  const int q0 = (blockIdx.y * 8 + wave) * 16;
  const int b = nh / NHEAD, h = nh % NHEAD;
  const u16* Qh = Qb + (size_t)nh * S_LEN * HDIM;
  const u16* Kh = Kb + (size_t)nh * S_LEN * HDIM;
  const u16* Vh = Vt + (size_t)nh * HDIM * S_LEN;
  const float* biasH = bias + (size_t)nh * S_LEN * S_LEN;
  const unsigned char* maskB = mask + (size_t)b * S_LEN;
  const int l15 = lane & 15;
  const int rbase = (lane >> 4) << 3;   // 0 or 8: which 8 query rows this lane covers

  const v16bf qa0 = ld_a(Qh, HDIM, q0 + l15, 0, lane);
  const v16bf qa1 = ld_a(Qh, HDIM, q0 + l15, 32, lane);

  float mi[8], li[8];
  v8f acc[4];
#pragma unroll
  for (int r = 0; r < 8; ++r) { mi[r] = -3.0e38f; li[r] = 0.f; }
#pragma unroll
  for (int t = 0; t < 4; ++t) acc[t] = (v8f){0.f,0.f,0.f,0.f,0.f,0.f,0.f,0.f};

  u16* P = &Pl[wave][0];

  for (int kt = 0; kt < S_LEN; kt += 32) {
    // scores: 16x32 tile as two 16x16 D-fragments
    v8f s0 = {0.f,0.f,0.f,0.f,0.f,0.f,0.f,0.f};
    v8f s1 = {0.f,0.f,0.f,0.f,0.f,0.f,0.f,0.f};
    {
      v16bf kb0 = ld_b(Kh, HDIM, kt + l15,       0, lane);
      v16bf kb1 = ld_b(Kh, HDIM, kt + l15,      32, lane);
      v16bf kb2 = ld_b(Kh, HDIM, kt + 16 + l15,  0, lane);
      v16bf kb3 = ld_b(Kh, HDIM, kt + 16 + l15, 32, lane);
      s0 = wmma_bf16(qa0, kb0, s0);
      s0 = wmma_bf16(qa1, kb1, s0);
      s1 = wmma_bf16(qa0, kb2, s1);
      s1 = wmma_bf16(qa1, kb3, s1);
    }

    // bias + key-padding mask (dominant HBM stream: prefetch next chunk)
    const int c0 = kt + l15, c1 = c0 + 16;
    const float mk0 = maskB[c0] ? -3.0e38f : 0.f;
    const float mk1 = maskB[c1] ? -3.0e38f : 0.f;
    const float* bp = biasH + (size_t)(q0 + rbase) * S_LEN;
    if (kt + 32 < S_LEN)
      __builtin_prefetch(bp + kt + 32, 0, 1);
#pragma unroll
    for (int r = 0; r < 8; ++r) {
      s0[r] += bp[(size_t)r * S_LEN + c0] + mk0;
      s1[r] += bp[(size_t)r * S_LEN + c1] + mk1;
    }

    // online softmax (row reductions across 16 lanes of this half-wave)
#pragma unroll
    for (int r = 0; r < 8; ++r) {
      float v = fmaxf(s0[r], s1[r]);
      v = fmaxf(v, __shfl_xor(v, 1));
      v = fmaxf(v, __shfl_xor(v, 2));
      v = fmaxf(v, __shfl_xor(v, 4));
      v = fmaxf(v, __shfl_xor(v, 8));
      const float nm   = fmaxf(mi[r], v);
      const float corr = __expf(mi[r] - nm);
      mi[r] = nm;
      const float p0 = __expf(s0[r] - nm);
      const float p1 = __expf(s1[r] - nm);
      float rs = p0 + p1;
      rs += __shfl_xor(rs, 1);
      rs += __shfl_xor(rs, 2);
      rs += __shfl_xor(rs, 4);
      rs += __shfl_xor(rs, 8);
      li[r] = li[r] * corr + rs;
      acc[0][r] *= corr; acc[1][r] *= corr; acc[2][r] *= corr; acc[3][r] *= corr;
      // stage P (bf16) into LDS row-major [16][32]
      P[(rbase + r) * 32 + l15]      = f2bf(p0);
      P[(rbase + r) * 32 + 16 + l15] = f2bf(p1);
    }
    // within-wave LDS write->read ordering (CDNA5 split counter)
    asm volatile("s_wait_dscnt 0" ::: "memory");

    // reload P in A-fragment layout, accumulate O += P x V
    const v16bf pa = ld_a(P, 32, l15, 0, lane);
#pragma unroll
    for (int dt = 0; dt < 4; ++dt) {
      v16bf vb = ld_b(Vh, S_LEN, dt * 16 + l15, kt, lane);
      acc[dt] = wmma_bf16(pa, vb, acc[dt]);
    }
  }

  // epilogue: normalize and scatter into attn [token][e] bf16
#pragma unroll
  for (int r = 0; r < 8; ++r) {
    const float inv = 1.0f / li[r];
    const int m = (q0 + rbase + r) * BATCH + b;  // token row = s*BATCH + b
#pragma unroll
    for (int dt = 0; dt < 4; ++dt)
      attn[(size_t)m * EMB + h * HDIM + dt * 16 + l15] = f2bf(acc[dt][r] * inv);
  }
}

// ---------------- kernel 4: output projection --------------------
__global__ __launch_bounds__(256) void k_out(
    const u16* __restrict__ A, const u16* __restrict__ Wo,
    const float* __restrict__ bo, float* __restrict__ out) {
  const int lane = threadIdx.x & 31, wave = threadIdx.x >> 5;
  const int n0 = blockIdx.x * 16;
  const int m0 = (blockIdx.y * 8 + wave) * 16;
  const int l15 = lane & 15;
  const int rA = m0 + l15, cB = n0 + l15;
  v8f c = {0.f,0.f,0.f,0.f,0.f,0.f,0.f,0.f};
  for (int k0 = 0; k0 < EMB; k0 += 32) {
    v16bf a = ld_a(A,  EMB, rA, k0, lane);
    v16bf b = ld_b(Wo, EMB, cB, k0, lane);
    c = wmma_bf16(a, b, c);
  }
  const float bval = bo[cB];
  const int rbase = m0 + ((lane >> 4) << 3);
#pragma unroll
  for (int r = 0; r < 8; ++r)
    out[(size_t)(rbase + r) * EMB + cB] = c[r] + bval;
}

// ---------------- launcher --------------------------------------
extern "C" void kernel_launch(void* const* d_in, const int* in_sizes, int n_in,
                              void* d_out, int out_size, void* d_ws, size_t ws_size,
                              hipStream_t stream) {
  const float* query      = (const float*)d_in[0];
  const float* attn_bias  = (const float*)d_in[1];
  const unsigned char* km = (const unsigned char*)d_in[2];  // bool mask
  const float* Wq = (const float*)d_in[3];
  const float* bq = (const float*)d_in[4];
  const float* Wk = (const float*)d_in[5];
  const float* bk = (const float*)d_in[6];
  const float* Wv = (const float*)d_in[7];
  const float* bv = (const float*)d_in[8];
  const float* Wo = (const float*)d_in[9];
  const float* bo = (const float*)d_in[10];
  float* out = (float*)d_out;

  // carve bf16 workspace
  char* ws = (char*)d_ws;
  size_t off = 0;
  auto carve = [&](size_t bytes) -> u16* {
    u16* p = (u16*)(ws + off);
    off += (bytes + 255) & ~(size_t)255;
    return p;
  };
  const size_t nX = (size_t)M_TOT * EMB;          // 3,145,728
  const size_t nW = (size_t)EMB * EMB;            //   589,824
  const size_t nH = (size_t)NH_TOT * S_LEN * HDIM;// 3,145,728
  u16* Xb  = carve(nX * 2);
  u16* Wqb = carve(nW * 2);
  u16* Wkb = carve(nW * 2);
  u16* Wvb = carve(nW * 2);
  u16* Wob = carve(nW * 2);
  u16* Qb  = carve(nH * 2);
  u16* Kb  = carve(nH * 2);
  u16* Vt  = carve(nH * 2);
  u16* At  = carve(nX * 2);
  (void)ws_size; (void)in_sizes; (void)n_in; (void)out_size;

  // 1) converts
  k_cvt<<<2048, 256, 0, stream>>>(query, Xb, (int)nX);
  k_cvt<<<1024, 256, 0, stream>>>(Wq, Wqb, (int)nW);
  k_cvt<<<1024, 256, 0, stream>>>(Wk, Wkb, (int)nW);
  k_cvt<<<1024, 256, 0, stream>>>(Wv, Wvb, (int)nW);
  k_cvt<<<1024, 256, 0, stream>>>(Wo, Wob, (int)nW);

  // 2) QKV projections: 48 N-tiles x 32 blocks of 8 M-tile waves x 3 proj
  dim3 g2(EMB / 16, M_TOT / (16 * 8), 3);
  k_qkv<<<g2, 256, 0, stream>>>(Xb, Wqb, Wkb, Wvb, bq, bk, bv, Qb, Kb, Vt);

  // 3) flash attention: 24 heads x (128 q-tiles / 8 waves)
  dim3 g3(NH_TOT, S_LEN / (16 * 8));
  k_attn<<<g3, 256, 0, stream>>>(Qb, Kb, Vt, attn_bias, km, At);

  // 4) output projection -> fp32
  dim3 g4(EMB / 16, M_TOT / (16 * 8));
  k_out<<<g4, 256, 0, stream>>>(At, Wob, bo, out);
}
